// ExtractTensorPatches_64029372449171
// MI455X (gfx1250) — compile-verified, matrix operand order checked by reading, not run on hardware
//
#include <hip/hip_runtime.h>

typedef __attribute__((ext_vector_type(16))) _Float16 v16h;
typedef __attribute__((ext_vector_type(8)))  float    v8f;
typedef __attribute__((ext_vector_type(4)))  float    f32x4;

namespace {
constexpr int   kWin     = 16;
constexpr int   kStride  = 8;
constexpr int   kH       = 256;
constexpr int   kW       = 256;
constexpr int   kC       = 32;
constexpr int   kB       = 4;
constexpr int   kHo      = 31;              // (256-16)/8 + 1
constexpr int   kL       = kHo * kHo;       // 961 patch locations
constexpr int   kPatches = kB * kL * kC;    // 123008 (exactly divisible by 8)
constexpr float kEps     = 1e-6f;
}

// One wave32 per (b, l, c) patch.
//   out[i][j] = P[i][j] + eps * sum(P)
// sum(P): v_wmma_f32_16x16x32_f16 with A = P (f16, K zero-padded 16->32),
// B = ones  =>  D element v at lane λ = rowsum(v + 8*(λ>=16)).
__global__ __launch_bounds__(256, 1) void extract_patches_wmma(
    const float* __restrict__ x, float* __restrict__ out)
{
  const int lane  = threadIdx.x & 31;
  const int wave  = threadIdx.x >> 5;
  const int patch = blockIdx.x * 8 + wave;          // 0 .. kPatches-1

  // patch = ((b*kL + l) * kC) + c   (matches output layout (B, L, C, 16, 16))
  const int c  = patch & (kC - 1);
  const int t  = patch >> 5;                        // b*kL + l
  const int l  = t % kL;
  const int b  = t / kL;
  const int ph = l / kHo;
  const int pw = l - ph * kHo;

  const float* xp = x + ((size_t)(b * kC + c) * kH + ph * kStride) * kW
                      + pw * kStride;

  // ---------------- WMMA patch reduction: T = P x Ones ----------------
  // A-layout (f16 16x32): lane λ -> row m = λ%16, K-cols 8*hi..8*hi+7
  const int m  = lane & 15;
  const int hi = lane >> 4;
  const float* ap = xp + m * kW + hi * 8;
  f32x4 a0 = *(const f32x4*)(ap);
  f32x4 a1 = *(const f32x4*)(ap + 4);

  v16h A = {};                                      // elements 8..15 stay 0 (K>=16 pad)
  A[0] = (_Float16)a0.x; A[1] = (_Float16)a0.y;
  A[2] = (_Float16)a0.z; A[3] = (_Float16)a0.w;
  A[4] = (_Float16)a1.x; A[5] = (_Float16)a1.y;
  A[6] = (_Float16)a1.z; A[7] = (_Float16)a1.w;

  v16h Bones;
#pragma unroll
  for (int i = 0; i < 16; ++i) Bones[i] = (_Float16)1.0f;

  v8f Cz = {};
  // D[m][n] = rowsum(m); f32 accumulation of f16 products.
  v8f T = __builtin_amdgcn_wmma_f32_16x16x32_f16(
      /*neg_a=*/false, A, /*neg_b=*/false, Bones,
      /*c_mod=*/(short)0, Cz, /*reuse_a=*/false, /*reuse_b=*/false);

  // lanes 0-15 hold rowsums 0..7, lanes 16-31 hold rowsums 8..15
  float part = ((T[0] + T[1]) + (T[2] + T[3])) + ((T[4] + T[5]) + (T[6] + T[7]));
  float S    = part + __shfl_xor(part, 16, 32);     // full patch sum in every lane
  const float add = kEps * S;

  // ---------------- streaming path: out = P + eps*S ----------------
  // lane λ owns flat patch elements [4λ, 4λ+3] and [128+4λ, 128+4λ+3]
  const int i0 = lane >> 2;
  const int j0 = (lane & 3) << 2;
  f32x4 p0 = *(const f32x4*)(xp + i0 * kW + j0);
  f32x4 p1 = *(const f32x4*)(xp + (i0 + 8) * kW + j0);
  p0.x += add; p0.y += add; p0.z += add; p0.w += add;
  p1.x += add; p1.y += add; p1.z += add; p1.w += add;

  // Coalesced, nontemporal stores (write-once 126MB stream: don't evict the
  // 32MB input from the 192MB L2, which services all overlapping re-reads).
  float* op = out + (size_t)patch * (kWin * kWin);
  __builtin_nontemporal_store(p0, (f32x4*)(op + 4 * lane));
  __builtin_nontemporal_store(p1, (f32x4*)(op + 128 + 4 * lane));
}

extern "C" void kernel_launch(void* const* d_in, const int* /*in_sizes*/, int /*n_in*/,
                              void* d_out, int /*out_size*/, void* /*d_ws*/, size_t /*ws_size*/,
                              hipStream_t stream) {
  const float* x   = (const float*)d_in[0];
  float*       out = (float*)d_out;
  dim3 block(256);                     // 8 wave32s per block -> 8 patches
  dim3 grid(kPatches / 8);             // 15376 blocks
  hipLaunchKernelGGL(extract_patches_wmma, grid, block, 0, stream, x, out);
}
